// TransformerBlock_63256278335970
// MI455X (gfx1250) — matrix operand order repaired and checked by reading, not compiled, not run on hardware
//
#include <hip/hip_runtime.h>
#include <hip/hip_bf16.h>
#include <math.h>

// ---------------------------------------------------------------------------
// Transformer block for MI455X (gfx1250): bf16 WMMA everywhere, flash attn,
// async global->LDS staging (ASYNCcnt) with a peeled double-buffer pipeline.
// Shapes fixed by the reference: B=1, T=4096, D=768, H=12, dk=64, DFF=3072.
// ---------------------------------------------------------------------------

#define T_SEQ 4096
#define DM    768
#define DFF   3072
#define NH    12
#define DKH   64

typedef __attribute__((ext_vector_type(16))) __bf16 v16bf;
typedef __attribute__((ext_vector_type(8)))  __bf16 v8bf;
typedef __attribute__((ext_vector_type(8)))  float  v8f;

// ---------------------------------------------------------------------------
// Fragment loaders (CDNA5 16-bit WMMA VGPR layouts, wave32)
// A (16x32, MxK): lane m=lane&15; lanes 0-15 hold K {0..7,16..23},
//                 lanes 16-31 hold K {8..15,24..31}. v16bf elem i -> VGPR i/2.
// B (32x16, KxN): lane n=lane&15; same K split as A ({0..7,16..23} / {8..15,24..31}).
// C/D (16x16 f32): reg r -> row r + 8*(lane>=16), col = lane&15.
// ---------------------------------------------------------------------------
__device__ __forceinline__ v16bf frag_cat(v8bf lo, v8bf hi) {
  v16bf r;
#pragma unroll
  for (int i = 0; i < 8; ++i) { r[i] = lo[i]; r[i + 8] = hi[i]; }
  return r;
}

// src points at tile origin (row M0, col K0) of a row-major bf16 matrix, ld in elems
__device__ __forceinline__ v16bf load_frag_A(const __bf16* __restrict__ src, int ld, int lane) {
  const int m   = lane & 15;
  const int off = (lane >> 4) ? 8 : 0;
  const __bf16* p = src + (size_t)m * ld + off;
  v8bf lo = *(const v8bf*)(p);        // K = off .. off+7
  v8bf hi = *(const v8bf*)(p + 16);   // K = off+16 .. off+23
  return frag_cat(lo, hi);
}

// srcT points at tile origin (row N0, col K0) of B^T stored [N][K] row-major
__device__ __forceinline__ v16bf load_frag_B(const __bf16* __restrict__ srcT, int ld, int lane) {
  const int n   = lane & 15;
  const int off = (lane >> 4) ? 8 : 0;
  const __bf16* p = srcT + (size_t)n * ld + off;
  v8bf lo = *(const v8bf*)(p);        // K = off .. off+7
  v8bf hi = *(const v8bf*)(p + 16);   // K = off+16 .. off+23
  return frag_cat(lo, hi);
}

__device__ __forceinline__ v8f wmma_bf16(v16bf a, v16bf b, v8f c) {
  return __builtin_amdgcn_wmma_f32_16x16x32_bf16(false, a, false, b, (short)0, c,
                                                 false, false);
}

// gfx1250 async global->LDS copy (tracked by ASYNCcnt). ldsOff is the
// allocation-relative LDS byte offset (low 32 bits of the generic address,
// per the ISA flat-aperture rules).
__device__ __forceinline__ void async_load_b128(void* ldsDst, const void* gSrc) {
  unsigned ldsOff = (unsigned)(uintptr_t)ldsDst;
  asm volatile("global_load_async_to_lds_b128 %0, %1, off"
               :: "v"(ldsOff), "v"(gSrc)
               : "memory");
}
__device__ __forceinline__ void wait_asynccnt0() {
  asm volatile("s_wait_asynccnt 0x0" ::: "memory");
}

// ---------------------------------------------------------------------------
// Conversion / transpose kernels (one-time setup per launch)
// ---------------------------------------------------------------------------
__global__ void cvt_f32_to_bf16(const float* __restrict__ in, __bf16* __restrict__ out, int n) {
  int i = blockIdx.x * 256 + threadIdx.x;
  if (i < n) out[i] = (__bf16)in[i];
}

// W[K][N] f32 -> Wt[N][K] bf16 (K,N multiples of 32)
__global__ __launch_bounds__(256)
void transpose_to_bf16(const float* __restrict__ W, __bf16* __restrict__ Wt, int K, int N) {
  __shared__ float tile[32][33];
  const int tx = threadIdx.x & 31, ty = threadIdx.x >> 5;
  const int k0 = blockIdx.x * 32, n0 = blockIdx.y * 32;
#pragma unroll
  for (int i = 0; i < 4; ++i)
    tile[ty + i * 8][tx] = W[(size_t)(k0 + ty + i * 8) * N + (n0 + tx)];
  __syncthreads();
#pragma unroll
  for (int i = 0; i < 4; ++i)
    Wt[(size_t)(n0 + ty + i * 8) * K + (k0 + tx)] = (__bf16)tile[tx][ty + i * 8];
}

// ---------------------------------------------------------------------------
// GEMM: out[M][N] = act(A[M][K]_bf16 x Wt[N][K]_bf16 + bias[N]) * outScale
// Block = 256 thr = 8 waves; block tile 256(M) x 64(N); wave tile 32 x 64.
// B^T tile double-buffered in LDS via async global->LDS copies with a peeled
// pipeline: steady-state loop is wait -> barrier -> async issue -> 8 WMMAs,
// no conditionals. Epilogue fully specialized by template flags.
// ---------------------------------------------------------------------------
template <int RELU, int TRANSB, int HASF, int HASB>
__global__ __launch_bounds__(256)
void gemm_bf16_wmma(const __bf16* __restrict__ A, const __bf16* __restrict__ Bt,
                    const float* __restrict__ bias,
                    float* __restrict__ outF, __bf16* __restrict__ outB,
                    int M, int N, int K, float outScale) {
  __shared__ __bf16 sB[2][64 * 32];              // 2 x 4 KB
  const int tid  = threadIdx.x;
  const int lane = tid & 31;
  const int wave = tid >> 5;
  const int m0 = blockIdx.x * 256 + wave * 32;
  const int n0 = blockIdx.y * 64;

  v8f acc[2][4] = {};

  // cooperative staging: 256 threads x 16B = 64(N) x 32(K) bf16 tile
  const int br = tid >> 2;                       // 0..63  (N within tile)
  const int bc = (tid & 3) * 8;                  // 0,8,16,24 (K within tile)
  const __bf16* gB = Bt + (size_t)(n0 + br) * K + bc;

  const __bf16* aRow0 = A + (size_t)m0 * K;
  const __bf16* aRow1 = A + (size_t)(m0 + 16) * K;

  const int nk = K / 32;
  async_load_b128(&sB[0][br * 32 + bc], gB);

  auto compute = [&](int kt) {
    const int k0 = kt * 32;
    v16bf a0 = load_frag_A(aRow0 + k0, K, lane);
    v16bf a1 = load_frag_A(aRow1 + k0, K, lane);
    const __bf16* sbuf = &sB[kt & 1][0];
#pragma unroll
    for (int j = 0; j < 4; ++j) {
      v16bf b = load_frag_B(sbuf + j * 16 * 32, 32, lane);
      acc[0][j] = wmma_bf16(a0, b, acc[0][j]);
      acc[1][j] = wmma_bf16(a1, b, acc[1][j]);
    }
  };

  for (int kt = 0; kt < nk - 1; ++kt) {
    wait_asynccnt0();
    __syncthreads();
    async_load_b128(&sB[(kt + 1) & 1][br * 32 + bc], gB + (kt + 1) * 32);
    compute(kt);
  }
  wait_asynccnt0();
  __syncthreads();
  compute(nk - 1);

  const int col  = lane & 15;
  const int rofs = (lane >> 4) * 8;
#pragma unroll
  for (int g = 0; g < 2; ++g) {
#pragma unroll
    for (int j = 0; j < 4; ++j) {
      const int n  = n0 + j * 16 + col;
      const float bv = bias[n];
#pragma unroll
      for (int r = 0; r < 8; ++r) {
        const int m = m0 + g * 16 + rofs + r;
        float v = acc[g][j][r] + bv;
        if (RELU) v = fmaxf(v, 0.0f);
        v *= outScale;
        if (HASF) outF[(size_t)m * N + n] = v;
        if (HASB) {
          if (TRANSB) outB[(size_t)n * M + m] = (__bf16)v;   // V^T layout [N][M]
          else        outB[(size_t)m * N + n] = (__bf16)v;
        }
      }
    }
  }
}

// ---------------------------------------------------------------------------
// Flash attention forward, causal. Q[T][768] (bf16, pre-scaled by 1/sqrt(dk)),
// K[T][768] (bf16), Vt[768][T] (bf16, head rows h*64..h*64+63), O[T][768] bf16.
// Block = 8 waves; wave owns one 16-row query tile; key tiles of 32.
// ---------------------------------------------------------------------------
__global__ __launch_bounds__(256)
void flash_attn_wmma(const __bf16* __restrict__ Q, const __bf16* __restrict__ Kb,
                     const __bf16* __restrict__ Vt, __bf16* __restrict__ O) {
  __shared__ __bf16 sP[8][16 * 32];              // per-wave P tile, 8 KB total
  const int tid  = threadIdx.x;
  const int lane = tid & 31;
  const int wave = tid >> 5;
  const int h  = blockIdx.y;
  const int q0 = (blockIdx.x * 8 + wave) * 16;
  const int col  = lane & 15;
  const int rofs = (lane >> 4) * 8;

  const v16bf aq0 = load_frag_A(Q + (size_t)q0 * DM + h * DKH + 0,  DM, lane);
  const v16bf aq1 = load_frag_A(Q + (size_t)q0 * DM + h * DKH + 32, DM, lane);

  v8f o[4] = {};
  float mi[8], li[8];
#pragma unroll
  for (int r = 0; r < 8; ++r) { mi[r] = -__builtin_inff(); li[r] = 0.0f; }

  for (int kb = 0; kb < q0 + 16; kb += 32) {
    // prefetch next key tile while this one computes
    if (kb + 32 < q0 + 16)
      __builtin_prefetch(Kb + (size_t)(kb + 32) * DM + h * DKH, 0, 1);

    // S tile: 16(q) x 32(k), two 16x16 D tiles, each reduced over dk=64
    v8f s0 = {}, s1 = {};
    {
      const __bf16* kt0 = Kb + (size_t)(kb)      * DM + h * DKH;
      const __bf16* kt1 = Kb + (size_t)(kb + 16) * DM + h * DKH;
      s0 = wmma_bf16(aq0, load_frag_B(kt0,      DM, lane), s0);
      s0 = wmma_bf16(aq1, load_frag_B(kt0 + 32, DM, lane), s0);
      s1 = wmma_bf16(aq0, load_frag_B(kt1,      DM, lane), s1);
      s1 = wmma_bf16(aq1, load_frag_B(kt1 + 32, DM, lane), s1);
    }

    // online softmax per row (rows live in regs r, cols across 16-lane groups)
#pragma unroll
    for (int r = 0; r < 8; ++r) {
      const int qrow = q0 + rofs + r;
      float v0 = s0[r];                          // Q pre-scaled by 1/sqrt(dk)
      float v1 = s1[r];
      if (kb + col      > qrow) v0 = -__builtin_inff();
      if (kb + 16 + col > qrow) v1 = -__builtin_inff();
      float mx = fmaxf(v0, v1);
#pragma unroll
      for (int d = 1; d < 16; d <<= 1) mx = fmaxf(mx, __shfl_xor(mx, d, 32));
      const float mn = fmaxf(mi[r], mx);
      const float a  = __expf(mi[r] - mn);
      const float p0 = __expf(v0 - mn);
      const float p1 = __expf(v1 - mn);
      float sum = p0 + p1;
#pragma unroll
      for (int d = 1; d < 16; d <<= 1) sum += __shfl_xor(sum, d, 32);
      li[r] = li[r] * a + sum;
      mi[r] = mn;
#pragma unroll
      for (int t = 0; t < 4; ++t) o[t][r] *= a;
      __bf16* pr = &sP[wave][(rofs + r) * 32];
      pr[col]      = (__bf16)p0;
      pr[16 + col] = (__bf16)p1;
    }

    // P (16x32) x V tile (32x64): A-frag re-read from LDS (D->A relayout)
    const v16bf ap = load_frag_A(&sP[wave][0], 32, lane);
#pragma unroll
    for (int t = 0; t < 4; ++t) {
      const __bf16* vtt = Vt + (size_t)(h * DKH + t * 16) * T_SEQ + kb;
      o[t] = wmma_bf16(ap, load_frag_B(vtt, T_SEQ, lane), o[t]);
    }
  }

#pragma unroll
  for (int r = 0; r < 8; ++r) {
    const float inv = 1.0f / li[r];
    const int m = q0 + rofs + r;
#pragma unroll
    for (int t = 0; t < 4; ++t)
      O[(size_t)m * DM + h * DKH + t * 16 + col] = (__bf16)(o[t][r] * inv);
  }
}

// ---------------------------------------------------------------------------
// Fused residual add + LayerNorm. One block per row, D = 768 = 3*256.
// ---------------------------------------------------------------------------
__global__ __launch_bounds__(256)
void add_layernorm(const float* __restrict__ X, const float* __restrict__ Y,
                   const float* __restrict__ g, const float* __restrict__ b,
                   float* __restrict__ outF, __bf16* __restrict__ outB) {
  __shared__ float red[16];
  const int row  = blockIdx.x;
  const int lane = threadIdx.x & 31, wave = threadIdx.x >> 5;
  const float* x = X + (size_t)row * DM;
  const float* y = Y + (size_t)row * DM;

  float vals[3], s = 0.0f, s2 = 0.0f;
#pragma unroll
  for (int i = 0; i < 3; ++i) {
    const int idx = threadIdx.x + i * 256;
    const float v = x[idx] + y[idx];
    vals[i] = v; s += v; s2 += v * v;
  }
#pragma unroll
  for (int d = 1; d < 32; d <<= 1) { s += __shfl_xor(s, d, 32); s2 += __shfl_xor(s2, d, 32); }
  if (lane == 0) { red[wave] = s; red[8 + wave] = s2; }
  __syncthreads();
  s = 0.0f; s2 = 0.0f;
#pragma unroll
  for (int w = 0; w < 8; ++w) { s += red[w]; s2 += red[8 + w]; }
  const float mu  = s / (float)DM;
  const float var = s2 / (float)DM - mu * mu;
  const float rs  = rsqrtf(var + 1e-5f);
#pragma unroll
  for (int i = 0; i < 3; ++i) {
    const int idx = threadIdx.x + i * 256;
    const float v = (vals[i] - mu) * rs * g[idx] + b[idx];
    outF[(size_t)row * DM + idx] = v;
    if (outB) outB[(size_t)row * DM + idx] = (__bf16)v;
  }
}

// ---------------------------------------------------------------------------
// Host-side orchestration
// ---------------------------------------------------------------------------
extern "C" void kernel_launch(void* const* d_in, const int* in_sizes, int n_in,
                              void* d_out, int out_size, void* d_ws, size_t ws_size,
                              hipStream_t stream) {
  const float* x   = (const float*)d_in[0];
  const float* Wq  = (const float*)d_in[1];  const float* bq = (const float*)d_in[2];
  const float* Wk  = (const float*)d_in[3];  const float* bk = (const float*)d_in[4];
  const float* Wv  = (const float*)d_in[5];  const float* bv = (const float*)d_in[6];
  const float* Wo  = (const float*)d_in[7];  const float* bo = (const float*)d_in[8];
  const float* W1  = (const float*)d_in[9];  const float* b1 = (const float*)d_in[10];
  const float* W2  = (const float*)d_in[11]; const float* b2 = (const float*)d_in[12];
  const float* g1  = (const float*)d_in[13]; const float* be1 = (const float*)d_in[14];
  const float* g2  = (const float*)d_in[15]; const float* be2 = (const float*)d_in[16];
  float* out = (float*)d_out;
  (void)in_sizes; (void)n_in; (void)out_size; (void)ws_size;

  char* ws = (char*)d_ws;
  size_t off = 0;
  auto alloc = [&](size_t bytes) -> void* {
    void* p = ws + off;
    off = (off + bytes + 255) & ~(size_t)255;
    return p;
  };

  __bf16* WqT = (__bf16*)alloc((size_t)DM * DM * 2);
  __bf16* WkT = (__bf16*)alloc((size_t)DM * DM * 2);
  __bf16* WvT = (__bf16*)alloc((size_t)DM * DM * 2);
  __bf16* WoT = (__bf16*)alloc((size_t)DM * DM * 2);
  __bf16* W1T = (__bf16*)alloc((size_t)DM * DFF * 2);
  __bf16* W2T = (__bf16*)alloc((size_t)DFF * DM * 2);
  __bf16* xB  = (__bf16*)alloc((size_t)T_SEQ * DM * 2);
  __bf16* qB  = (__bf16*)alloc((size_t)T_SEQ * DM * 2);
  __bf16* kB  = (__bf16*)alloc((size_t)T_SEQ * DM * 2);
  __bf16* vT  = (__bf16*)alloc((size_t)DM * T_SEQ * 2);     // [768][T]
  __bf16* oB  = (__bf16*)alloc((size_t)T_SEQ * DM * 2);
  float*  atF = (float*) alloc((size_t)T_SEQ * DM * 4);
  float*  hF  = (float*) alloc((size_t)T_SEQ * DM * 4);
  __bf16* hB  = (__bf16*)alloc((size_t)T_SEQ * DM * 2);
  __bf16* ffB = (__bf16*)alloc((size_t)T_SEQ * DFF * 2);
  float*  f2F = (float*) alloc((size_t)T_SEQ * DM * 4);

  // 1) conversions
  {
    const int n = T_SEQ * DM;
    cvt_f32_to_bf16<<<(n + 255) / 256, 256, 0, stream>>>(x, xB, n);
  }
  transpose_to_bf16<<<dim3(DM / 32, DM / 32), 256, 0, stream>>>(Wq, WqT, DM, DM);
  transpose_to_bf16<<<dim3(DM / 32, DM / 32), 256, 0, stream>>>(Wk, WkT, DM, DM);
  transpose_to_bf16<<<dim3(DM / 32, DM / 32), 256, 0, stream>>>(Wv, WvT, DM, DM);
  transpose_to_bf16<<<dim3(DM / 32, DM / 32), 256, 0, stream>>>(Wo, WoT, DM, DM);
  transpose_to_bf16<<<dim3(DM / 32, DFF / 32), 256, 0, stream>>>(W1, W1T, DM, DFF);
  transpose_to_bf16<<<dim3(DFF / 32, DM / 32), 256, 0, stream>>>(W2, W2T, DFF, DM);

  const dim3 gProj(T_SEQ / 256, DM / 64);    // (16, 12)
  // 2) QKV projections. Q pre-scaled by 1/sqrt(dk); V stored transposed [768][T].
  gemm_bf16_wmma<0,0,0,1><<<gProj, 256, 0, stream>>>(xB, WqT, bq, nullptr, qB,
                                                     T_SEQ, DM, DM, 0.125f);
  gemm_bf16_wmma<0,0,0,1><<<gProj, 256, 0, stream>>>(xB, WkT, bk, nullptr, kB,
                                                     T_SEQ, DM, DM, 1.0f);
  gemm_bf16_wmma<0,1,0,1><<<gProj, 256, 0, stream>>>(xB, WvT, bv, nullptr, vT,
                                                     T_SEQ, DM, DM, 1.0f);

  // 3) flash attention (causal) -> oB bf16 [T][768]
  flash_attn_wmma<<<dim3(T_SEQ / 128, NH), 256, 0, stream>>>(qB, kB, vT, oB);

  // 4) output projection -> f32
  gemm_bf16_wmma<0,0,1,0><<<gProj, 256, 0, stream>>>(oB, WoT, bo, atF, nullptr,
                                                     T_SEQ, DM, DM, 1.0f);

  // 5) residual + LN1 -> hF (f32) + hB (bf16)
  add_layernorm<<<T_SEQ, 256, 0, stream>>>(x, atF, g1, be1, hF, hB);

  // 6) FFN
  gemm_bf16_wmma<1,0,0,1><<<dim3(T_SEQ / 256, DFF / 64), 256, 0, stream>>>(
      hB, W1T, b1, nullptr, ffB, T_SEQ, DFF, DM, 1.0f);
  gemm_bf16_wmma<0,0,1,0><<<gProj, 256, 0, stream>>>(ffB, W2T, b2, f2F, nullptr,
                                                     T_SEQ, DM, DFF, 1.0f);

  // 7) residual + LN2 -> d_out (f32)
  add_layernorm<<<T_SEQ, 256, 0, stream>>>(hF, f2F, g2, be2, out, nullptr);
}
